// EncGRU_51659866637020
// MI455X (gfx1250) — matrix operand order repaired and checked by reading, not compiled, not run on hardware
//
#include <hip/hip_runtime.h>

// Problem dims (match reference)
#define T_DIM 512
#define B_DIM 64
#define I_DIM 1024
#define H_DIM 1024
#define G_DIM (3 * H_DIM)   // 3072

typedef __bf16 bf16_t;
typedef __attribute__((ext_vector_type(16))) __bf16 bf16x16;
typedef __attribute__((ext_vector_type(8)))  float  f32x8;
typedef __attribute__((ext_vector_type(4)))  unsigned int u32x4;
typedef __attribute__((ext_vector_type(4)))  int  i32x4;

// addrspace-qualified v4i32 pointers for the async-copy builtin
typedef __attribute__((address_space(1))) i32x4 gi32x4;   // global
typedef __attribute__((address_space(3))) i32x4 li32x4;   // LDS

union FragB16 {
    bf16x16 v;
    u32x4   u[2];
};

// ---------------------------------------------------------------------------
// Async global->LDS 16-byte copy (ASYNCcnt-tracked, bypasses VGPRs).
// Guarded: falls back to a register copy (ds_store_b128) if builtin absent.
// ---------------------------------------------------------------------------
__device__ __forceinline__ void async_copy_b128(const void* g, void* l) {
#if __has_builtin(__builtin_amdgcn_global_load_async_to_lds_b128)
    __builtin_amdgcn_global_load_async_to_lds_b128(
        (gi32x4*)(unsigned long long)g,
        (li32x4*)(unsigned int)(unsigned long long)l,
        /*imm offset=*/0, /*cpol=*/0);
#else
    *(u32x4*)l = *(const u32x4*)g;
#endif
}

__device__ __forceinline__ void wait_async0() {
#if __has_builtin(__builtin_amdgcn_s_wait_asynccnt)
    __builtin_amdgcn_s_wait_asynccnt(0);
#else
    asm volatile("s_wait_asynccnt 0x0" ::: "memory");
#endif
}

// ---------------------------------------------------------------------------
// WMMA wrapper: D = A(16x32 bf16) x B(32x16 bf16) + C(16x16 f32)
// ---------------------------------------------------------------------------
__device__ __forceinline__ f32x8 wmma_bf16(bf16x16 a, bf16x16 b, f32x8 c) {
    return __builtin_amdgcn_wmma_f32_16x16x32_bf16(
        /*neg_a=*/false, a, /*neg_b=*/false, b,
        /*c_mod=*/(short)0, c, /*reuse_a=*/false, /*reuse_b=*/false);
}

// ---------------------------------------------------------------------------
// A fragment (16x32 bf16) from an LDS tile (row stride 72 bf16 = 144 B).
// ISA 16-bit A layout: lanes 0-15 -> K = kk+0..7 / kk+16..23; lanes 16-31 ->
// K = kk+8..15 / kk+24..31. Two 16B ds_load_b128 per lane.
// ---------------------------------------------------------------------------
__device__ __forceinline__ bf16x16 load_fragA_lds(const bf16_t* As, int mrow0,
                                                  int kk, int lane) {
    const int half = lane >> 4;
    const bf16_t* row = As + (mrow0 + (lane & 15)) * 72;
    FragB16 f;
    f.u[0] = *(const u32x4*)(row + kk +      half * 8);
    f.u[1] = *(const u32x4*)(row + kk + 16 + half * 8);
    return f.v;
}

// ---------------------------------------------------------------------------
// B fragment (32x16 bf16) from global row-major weights [N_rows, K].
// Lanes 0-15 -> K = kbase+0..7 / kbase+16..23, lanes 16-31 -> K = kbase+8..15 /
// kbase+24..31; per lane 16 bf16 from row (j0 + lane%16) = two global_load_b128.
// ---------------------------------------------------------------------------
__device__ __forceinline__ bf16x16 load_fragB_glb(const bf16_t* __restrict__ W, int ldw,
                                                  int j0, int kbase, int lane) {
    const int half = lane >> 4;
    const bf16_t* p = W + (size_t)(j0 + (lane & 15)) * ldw + kbase + half * 8;
    FragB16 f;
    f.u[0] = *(const u32x4*)(p);
    f.u[1] = *(const u32x4*)(p + 16);
    return f.v;
}

// ---------------------------------------------------------------------------
// Stage a 64x64 bf16 tile (8 KB) from global into an LDS buffer: 512 x b128
// async copies, 4 per thread (128 threads).
// ---------------------------------------------------------------------------
__device__ __forceinline__ void stage_tile(const bf16_t* __restrict__ gbase, int ldg,
                                           bf16_t* lbuf, int tid) {
    #pragma unroll
    for (int i = 0; i < 4; ++i) {
        int idx = i * 128 + tid;       // 0..511 b128 slots
        int r   = idx >> 3;            // row 0..63
        int c8  = (idx & 7) << 3;      // col 0..56 step 8
        async_copy_b128(gbase + (size_t)r * ldg + c8, lbuf + r * 72 + c8);
    }
}

// ---------------------------------------------------------------------------
// Prep kernels
// ---------------------------------------------------------------------------
__global__ void k_f32_to_bf16(const float* __restrict__ src, bf16_t* __restrict__ dst, int n) {
    int i = blockIdx.x * blockDim.x + threadIdx.x;
    if (i < n) dst[i] = (bf16_t)src[i];
}

__global__ void k_zero_f32(float* __restrict__ p, int n) {
    int i = blockIdx.x * blockDim.x + threadIdx.x;
    if (i < n) p[i] = 0.0f;
}

// ---------------------------------------------------------------------------
// Phase 1: x_proj[m, g] = sum_k seq[m,k] * W_ih[g,k] + b_ih[g]  (fp16 out)
//   Block: 128 threads (4 waves); each wave owns one 16x16 tile; block covers
//   64 M-rows x 16 N-cols. bf16 A tiles double-buffered through LDS with
//   async global->LDS copies overlapped against the WMMAs.
// ---------------------------------------------------------------------------
__global__ __launch_bounds__(128) void k_xproj(const bf16_t* __restrict__ seqb,
                                               const bf16_t* __restrict__ Wih,
                                               const float* __restrict__ bih,
                                               _Float16* __restrict__ xp) {
    __shared__ bf16_t As[2][64 * 72];
    const int tid  = threadIdx.x;
    const int lane = tid & 31;
    const int wave = tid >> 5;
    const int j0   = blockIdx.x * 16;
    const int m0   = blockIdx.y * 64;
    const bf16_t* arow = seqb + (size_t)m0 * I_DIM;

    f32x8 acc = {0.f, 0.f, 0.f, 0.f, 0.f, 0.f, 0.f, 0.f};

    const int NCH = I_DIM / 64;   // 16
    stage_tile(arow, I_DIM, As[0], tid);
    wait_async0();
    __syncthreads();
    for (int ci = 0; ci < NCH; ++ci) {
        const int cur = ci & 1;
        if (ci + 1 < NCH)
            stage_tile(arow + (ci + 1) * 64, I_DIM, As[cur ^ 1], tid);
        const int kc = ci * 64;
        #pragma unroll
        for (int kk = 0; kk < 64; kk += 32) {
            bf16x16 a = load_fragA_lds(As[cur], wave * 16, kk, lane);
            bf16x16 b = load_fragB_glb(Wih, I_DIM, j0, kc + kk, lane);
            acc = wmma_bf16(a, b, acc);
        }
        if (ci + 1 < NCH) wait_async0();
        __syncthreads();
    }

    const int half = lane >> 4;
    const int col  = j0 + (lane & 15);
    const float bias = bih[col];
    #pragma unroll
    for (int e = 0; e < 8; ++e) {
        int m = m0 + wave * 16 + e + half * 8;
        xp[(size_t)m * G_DIM + col] = (_Float16)(acc[e] + bias);
    }
}

// ---------------------------------------------------------------------------
// Phase 2: one GRU timestep (launched 512x sequentially).
//   Grid: H/16 = 64 blocks of 128 threads (4 waves). The bf16 shadow of
//   h_prev is staged async into double-buffered LDS; r/z/n accumulators share
//   each A tile. Epilogue fuses the gates and writes both fp32 h (d_out) and
//   the bf16 shadow for the next step's GEMM.
// ---------------------------------------------------------------------------
__global__ __launch_bounds__(128) void k_gru_step(const float* __restrict__ hprev,   // [B,H] fp32
                                                  const bf16_t* __restrict__ hbprev, // [B,H] bf16
                                                  const bf16_t* __restrict__ Whh,    // [3H,H] bf16
                                                  const float* __restrict__ bhh,     // [3H]
                                                  const _Float16* __restrict__ xp_t, // [B,3H]
                                                  float* __restrict__ hout,          // [B,H]
                                                  bf16_t* __restrict__ hbout) {      // [B,H] bf16
    __shared__ bf16_t As[2][64 * 72];
    const int tid  = threadIdx.x;
    const int lane = tid & 31;
    const int wave = tid >> 5;
    const int j0   = blockIdx.x * 16;

    f32x8 aR = {0.f, 0.f, 0.f, 0.f, 0.f, 0.f, 0.f, 0.f};
    f32x8 aZ = {0.f, 0.f, 0.f, 0.f, 0.f, 0.f, 0.f, 0.f};
    f32x8 aN = {0.f, 0.f, 0.f, 0.f, 0.f, 0.f, 0.f, 0.f};

    const bf16_t* Wr = Whh;
    const bf16_t* Wz = Whh + (size_t)H_DIM * H_DIM;
    const bf16_t* Wn = Whh + (size_t)2 * H_DIM * H_DIM;

    const int NCH = H_DIM / 64;   // 16
    stage_tile(hbprev, H_DIM, As[0], tid);
    wait_async0();
    __syncthreads();
    for (int ci = 0; ci < NCH; ++ci) {
        const int cur = ci & 1;
        if (ci + 1 < NCH)
            stage_tile(hbprev + (ci + 1) * 64, H_DIM, As[cur ^ 1], tid);
        const int kc = ci * 64;

        // Prefetch next K-chunk of the weight rows (global_prefetch_b8)
        if (kc + 64 < H_DIM)
            __builtin_prefetch(Wr + (size_t)(j0 + (lane & 15)) * H_DIM + kc + 64, 0, 0);

        #pragma unroll
        for (int kk = 0; kk < 64; kk += 32) {
            bf16x16 a  = load_fragA_lds(As[cur], wave * 16, kk, lane);
            bf16x16 bR = load_fragB_glb(Wr, H_DIM, j0, kc + kk, lane);
            bf16x16 bZ = load_fragB_glb(Wz, H_DIM, j0, kc + kk, lane);
            bf16x16 bN = load_fragB_glb(Wn, H_DIM, j0, kc + kk, lane);
            aR = wmma_bf16(a, bR, aR);
            aZ = wmma_bf16(a, bZ, aZ);
            aN = wmma_bf16(a, bN, aN);
        }
        if (ci + 1 < NCH) wait_async0();
        __syncthreads();
    }

    // Fused GRU gate epilogue (torch convention: gates laid out r, z, n)
    const int half = lane >> 4;
    const int col  = j0 + (lane & 15);
    const float biR = bhh[col];
    const float biZ = bhh[H_DIM + col];
    const float biN = bhh[2 * H_DIM + col];
    #pragma unroll
    for (int e = 0; e < 8; ++e) {
        int m = wave * 16 + e + half * 8;            // batch index
        const _Float16* xrow = xp_t + (size_t)m * G_DIM;
        float xr = (float)xrow[col];
        float xz = (float)xrow[H_DIM + col];
        float xn = (float)xrow[2 * H_DIM + col];
        float hr = aR[e] + biR;
        float hz = aZ[e] + biZ;
        float hn = aN[e] + biN;
        float r = 1.0f / (1.0f + __expf(-(xr + hr)));
        float z = 1.0f / (1.0f + __expf(-(xz + hz)));
        float n = tanhf(xn + r * hn);
        float hp = hprev[(size_t)m * H_DIM + col];
        float hv = (1.0f - z) * n + z * hp;
        hout[(size_t)m * H_DIM + col]  = hv;
        hbout[(size_t)m * H_DIM + col] = (bf16_t)hv;   // bf16 shadow for next GEMM
    }
}

// ---------------------------------------------------------------------------
// Launcher
// ---------------------------------------------------------------------------
extern "C" void kernel_launch(void* const* d_in, const int* in_sizes, int n_in,
                              void* d_out, int out_size, void* d_ws, size_t ws_size,
                              hipStream_t stream) {
    (void)in_sizes; (void)n_in; (void)out_size; (void)ws_size;

    const float* seq = (const float*)d_in[0];   // [T, B, I]
    const float* Wih = (const float*)d_in[1];   // [3H, I]
    const float* Whh = (const float*)d_in[2];   // [3H, H]
    const float* bih = (const float*)d_in[3];   // [3H]
    const float* bhh = (const float*)d_in[4];   // [3H]
    float* out = (float*)d_out;                 // [T, 1, B, H]

    // Workspace layout (~278 MB total)
    const size_t NWIH = (size_t)G_DIM * I_DIM;          // 3,145,728
    const size_t NWHH = (size_t)G_DIM * H_DIM;          // 3,145,728
    const size_t NSEQ = (size_t)T_DIM * B_DIM * I_DIM;  // 33,554,432
    const size_t NH0  = (size_t)B_DIM * H_DIM;          // 65,536
    char* ws = (char*)d_ws;
    size_t off = 0;
    bf16_t* wih_b = (bf16_t*)(ws + off); off += NWIH * sizeof(bf16_t);
    bf16_t* whh_b = (bf16_t*)(ws + off); off += NWHH * sizeof(bf16_t);
    bf16_t* seq_b = (bf16_t*)(ws + off); off += NSEQ * sizeof(bf16_t);
    float*  h0    = (float*) (ws + off); off += NH0  * sizeof(float);
    bf16_t* hbZ   = (bf16_t*)(ws + off); off += NH0  * sizeof(bf16_t);
    bf16_t* hb0   = (bf16_t*)(ws + off); off += NH0  * sizeof(bf16_t);
    bf16_t* hb1   = (bf16_t*)(ws + off); off += NH0  * sizeof(bf16_t);
    _Float16* xp  = (_Float16*)(ws + off);              // T*B*3H fp16 (~201 MB)

    // Prep: bf16 weights + bf16 seq, zeroed h0 (fp32 and bf16 shadow)
    k_f32_to_bf16<<<(int)((NWIH + 255) / 256), 256, 0, stream>>>(Wih, wih_b, (int)NWIH);
    k_f32_to_bf16<<<(int)((NWHH + 255) / 256), 256, 0, stream>>>(Whh, whh_b, (int)NWHH);
    k_f32_to_bf16<<<(int)((NSEQ + 255) / 256), 256, 0, stream>>>(seq, seq_b, (int)NSEQ);
    k_zero_f32<<<(int)((NH0 + 255) / 256), 256, 0, stream>>>(h0, (int)NH0);
    k_zero_f32<<<(int)((NH0 / 2 + 255) / 256), 256, 0, stream>>>((float*)hbZ, (int)(NH0 / 2));

    // Phase 1: input projection GEMM over all timesteps
    {
        dim3 grid(G_DIM / 16, (T_DIM * B_DIM) / 64);    // 192 x 512
        k_xproj<<<grid, 128, 0, stream>>>(seq_b, wih_b, bih, xp);
    }

    // Phase 2: sequential scan, one kernel per timestep (ping-pong bf16 h)
    const size_t BH = (size_t)B_DIM * H_DIM;
    const size_t BG = (size_t)B_DIM * G_DIM;
    for (int t = 0; t < T_DIM; ++t) {
        const float*  hprev  = (t == 0) ? h0  : (out + (size_t)(t - 1) * BH);
        const bf16_t* hbprev = (t == 0) ? hbZ : (((t - 1) & 1) ? hb1 : hb0);
        bf16_t*       hbout  = (t & 1) ? hb1 : hb0;
        k_gru_step<<<H_DIM / 16, 128, 0, stream>>>(hprev, hbprev, whh_b, bhh,
                                                   xp + (size_t)t * BG,
                                                   out + (size_t)t * BH, hbout);
    }
}
